// EGNNvelBaseline_56873956934479
// MI455X (gfx1250) — matrix operand order repaired and checked by reading, not compiled, not run on hardware
//
#include <hip/hip_runtime.h>
#include <stdint.h>

// ---------------- CDNA5 WMMA types ----------------
typedef __attribute__((ext_vector_type(16))) __bf16 v16bf;
typedef __attribute__((ext_vector_type(8)))  float  v8f;

#define NNODES 16000
#define NEDGES 304000
#define HID    128

// ---------------- scalar helpers ----------------
__device__ __forceinline__ unsigned short f2bf(float x) {
  unsigned int u = __float_as_uint(x);
  u += 0x7FFFu + ((u >> 16) & 1u);          // round-to-nearest-even
  return (unsigned short)(u >> 16);
}
__device__ __forceinline__ float siluf(float x) { return x / (1.0f + __expf(-x)); }
__device__ __forceinline__ float sigmf(float x) { return 1.0f / (1.0f + __expf(-x)); }

union AFrag { v16bf v; unsigned int u[8]; };
union BFrag { v16bf v; uint4 q[2]; };

// A-fragment (16x32 bf16, ISA 7.12.2 layout) from row-major bf16 [16][stride].
// lane half h: VGPR j -> K = kbase + h*8 + 2j ; VGPR 4+j -> K = kbase + h*8 + 16 + 2j
__device__ __forceinline__ v16bf load_afrag(const unsigned short* base, int stride,
                                            int lane, int kbase) {
  const int half = (lane >> 4) & 1, row = lane & 15;
  AFrag A;
  const unsigned short* p = base + row * stride + kbase + half * 8;
#pragma unroll
  for (int j = 0; j < 4; ++j) {
    A.u[j]     = *(const unsigned int*)(p + 2 * j);
    A.u[4 + j] = *(const unsigned int*)(p + 16 + 2 * j);
  }
  return A.v;
}

// A-fragment from f32 row-major LDS, scaled per-row (for m * sigmoid(att)).
__device__ __forceinline__ v16bf load_afrag_f32s(const float* base, int stride,
                                                 int lane, int kbase, const float* rowscale) {
  const int half = (lane >> 4) & 1, row = lane & 15;
  const float sc = rowscale[row];
  AFrag A;
  const float* p = base + row * stride + kbase + half * 8;
#pragma unroll
  for (int j = 0; j < 8; ++j) {
    const int off = (j < 4) ? 2 * j : 16 + 2 * (j - 4);
    unsigned short lo = f2bf(p[off] * sc);
    unsigned short hi = f2bf(p[off + 1] * sc);
    A.u[j] = ((unsigned int)hi << 16) | (unsigned int)lo;
  }
  return A.v;
}

// B-fragment from pre-packed weights: packed[((ks*8 + tile)*32 + lane)*16 ..]
__device__ __forceinline__ v16bf load_bfrag(const unsigned short* packed,
                                            int ks, int tile, int lane) {
  const unsigned short* p = packed + ((((ks * 8 + tile) * 32) + lane) << 4);
  BFrag B;
  B.q[0] = *(const uint4*)(p);
  B.q[1] = *(const uint4*)(p + 8);
  return B.v;
}

// one K-step, M=16: 8 N-tiles share one A fragment
__device__ __forceinline__ void wmma_step(v8f acc[8], v16bf A,
                                          const unsigned short* packed, int ks, int lane) {
#pragma unroll
  for (int t = 0; t < 8; ++t) {
    v16bf B = load_bfrag(packed, ks, t, lane);
    acc[t] = __builtin_amdgcn_wmma_f32_16x16x32_bf16(false, A, false, B,
                                                     (short)0, acc[t], false, false);
  }
}

// one K-step, M=32: two A fragments share every loaded B fragment
__device__ __forceinline__ void wmma_step2(v8f acc0[8], v8f acc1[8], v16bf A0, v16bf A1,
                                           const unsigned short* packed, int ks, int lane) {
#pragma unroll
  for (int t = 0; t < 8; ++t) {
    v16bf B = load_bfrag(packed, ks, t, lane);
    acc0[t] = __builtin_amdgcn_wmma_f32_16x16x32_bf16(false, A0, false, B,
                                                      (short)0, acc0[t], false, false);
    acc1[t] = __builtin_amdgcn_wmma_f32_16x16x32_bf16(false, A1, false, B,
                                                      (short)0, acc1[t], false, false);
  }
}

// ---------------- weight packing: f32 (K x 128) -> bf16 WMMA B fragments ----------------
// lane l: col n = tile*16 + (l&15); element i: K = ks*32 + (i<8 ? (l>>4)*8 + i : 16 + (l>>4)*8 + i-8)
__global__ void pack_kernel(const float* __restrict__ W, unsigned short* __restrict__ out,
                            int Kreal, int total) {
  int idx = blockIdx.x * blockDim.x + threadIdx.x;
  if (idx >= total) return;
  int i    = idx & 15;
  int lane = (idx >> 4) & 31;
  int tile = (idx >> 9) & 7;
  int ks   = idx >> 12;
  int n = tile * 16 + (lane & 15);
  int K = ks * 32 + ((lane >> 4) & 1) * 8 + (i < 8 ? i : 8 + i);
  float v = (K < Kreal) ? W[K * HID + n] : 0.0f;
  out[idx] = f2bf(v);
}

__global__ void zero_kernel(float* p, int n) {
  int i = blockIdx.x * blockDim.x + threadIdx.x;
  if (i < n) p[i] = 0.0f;
}

__global__ void copyloc_kernel(const float* __restrict__ src, float* __restrict__ dst, int n) {
  int i = blockIdx.x * blockDim.x + threadIdx.x;
  if (i < n) dst[i] = src[i];
}

// h = nodes @ We + be  (K=16, tiny -> VALU)
__global__ void embed_kernel(const float* __restrict__ nodes, const float* __restrict__ We,
                             const float* __restrict__ be,
                             float* __restrict__ h, unsigned short* __restrict__ hbf) {
  int idx = blockIdx.x * blockDim.x + threadIdx.x;
  if (idx >= NNODES * HID) return;
  int node = idx >> 7, col = idx & 127;
  float s = be[col];
#pragma unroll
  for (int k = 0; k < 16; ++k) s += nodes[node * 16 + k] * We[k * HID + col];
  h[idx] = s;
  hbf[idx] = f2bf(s);
}

// ---------------- fused edge pipeline (1 wave = 32 edges, M=32 tiling) ----------------
__global__ __launch_bounds__(32) void edge_kernel(
    const int* __restrict__ edges, const float* __restrict__ edge_attr,
    const float* __restrict__ loc, const unsigned short* __restrict__ hbf,
    const unsigned short* __restrict__ pW1, const float* __restrict__ b1,
    const unsigned short* __restrict__ pW2, const float* __restrict__ b2,
    const unsigned short* __restrict__ pWc1, const float* __restrict__ bc1,
    const float* __restrict__ Watt, const float* __restrict__ batt,
    const float* __restrict__ Wc2,
    float* __restrict__ agg_h, float* __restrict__ agg_c, float* __restrict__ cnt) {
  __shared__ __align__(16) unsigned short ein[32 * 288];   // padded e_in, bf16
  __shared__ __align__(16) unsigned short m1bf[32 * HID];  // silu(edge1) bf16
  __shared__ __align__(16) float mrow[32 * HID];           // m = silu(edge2)
  __shared__ __align__(16) float crow[32 * HID];           // silu(coord1)
  __shared__ int   srow[32], scol[32];
  __shared__ float sdiff[3][32], srad[32], satt[32], ssc[32];

  const int lane = threadIdx.x;
  const int e0 = blockIdx.x * 32;
  const v8f vzero = {0, 0, 0, 0, 0, 0, 0, 0};

  {
    int e = e0 + lane;
    int r = edges[e], c = edges[NEDGES + e];
    float d0 = loc[r * 3 + 0] - loc[c * 3 + 0];
    float d1 = loc[r * 3 + 1] - loc[c * 3 + 1];
    float d2 = loc[r * 3 + 2] - loc[c * 3 + 2];
    srow[lane] = r; scol[lane] = c;
    sdiff[0][lane] = d0; sdiff[1][lane] = d1; sdiff[2][lane] = d2;
    srad[lane] = d0 * d0 + d1 * d1 + d2 * d2;
  }
  __syncthreads();

  // build e_in = [h[row] | h[col] | radial | edge_attr | pad]
  for (int idx = lane; idx < 32 * 288; idx += 32) {
    int row = idx / 288;
    int K = idx - row * 288;
    unsigned short v;
    if (K < 128)        v = hbf[srow[row] * HID + K];
    else if (K < 256)   v = hbf[scol[row] * HID + (K - 128)];
    else if (K == 256)  v = f2bf(srad[row]);
    else if (K < 259)   v = f2bf(edge_attr[(e0 + row) * 2 + (K - 257)]);
    else                v = 0;
    ein[idx] = v;
  }
  __syncthreads();

  const int half = lane >> 4, cL = lane & 15;
  v8f acc0[8], acc1[8];

  // ---- GEMM1: e_in (32x288) @ W1 -> 32x128 ----
#pragma unroll
  for (int t = 0; t < 8; ++t) { acc0[t] = vzero; acc1[t] = vzero; }
#pragma unroll
  for (int ks = 0; ks < 9; ++ks) {
    v16bf A0 = load_afrag(ein,            288, lane, ks * 32);
    v16bf A1 = load_afrag(ein + 16 * 288, 288, lane, ks * 32);
    wmma_step2(acc0, acc1, A0, A1, pW1, ks, lane);
  }
#pragma unroll
  for (int t = 0; t < 8; ++t) {
    float bb = b1[t * 16 + cL];
#pragma unroll
    for (int v = 0; v < 8; ++v) {
      m1bf[(v + 8 * half) * HID + t * 16 + cL]        = f2bf(siluf(acc0[t][v] + bb));
      m1bf[(16 + v + 8 * half) * HID + t * 16 + cL]   = f2bf(siluf(acc1[t][v] + bb));
    }
  }
  __syncthreads();

  // ---- GEMM2: (32x128) @ W2 -> m ----
#pragma unroll
  for (int t = 0; t < 8; ++t) { acc0[t] = vzero; acc1[t] = vzero; }
#pragma unroll
  for (int ks = 0; ks < 4; ++ks) {
    v16bf A0 = load_afrag(m1bf,            HID, lane, ks * 32);
    v16bf A1 = load_afrag(m1bf + 16 * HID, HID, lane, ks * 32);
    wmma_step2(acc0, acc1, A0, A1, pW2, ks, lane);
  }
#pragma unroll
  for (int t = 0; t < 8; ++t) {
    float bb = b2[t * 16 + cL];
#pragma unroll
    for (int v = 0; v < 8; ++v) {
      mrow[(v + 8 * half) * HID + t * 16 + cL]      = siluf(acc0[t][v] + bb);
      mrow[(16 + v + 8 * half) * HID + t * 16 + cL] = siluf(acc1[t][v] + bb);
    }
  }
  __syncthreads();

  // ---- attention: satt = sigmoid(m @ Watt + batt); one row per lane ----
  {
    float s = batt[0];
    for (int k = 0; k < HID; ++k) s += mrow[lane * HID + k] * Watt[k];
    satt[lane] = sigmf(s);
  }
  __syncthreads();

  // ---- GEMM3: (m*att) @ Wc1 -> silu -> crow ----
#pragma unroll
  for (int t = 0; t < 8; ++t) { acc0[t] = vzero; acc1[t] = vzero; }
#pragma unroll
  for (int ks = 0; ks < 4; ++ks) {
    v16bf A0 = load_afrag_f32s(mrow,            HID, lane, ks * 32, satt);
    v16bf A1 = load_afrag_f32s(mrow + 16 * HID, HID, lane, ks * 32, satt + 16);
    wmma_step2(acc0, acc1, A0, A1, pWc1, ks, lane);
  }
#pragma unroll
  for (int t = 0; t < 8; ++t) {
    float bb = bc1[t * 16 + cL];
#pragma unroll
    for (int v = 0; v < 8; ++v) {
      crow[(v + 8 * half) * HID + t * 16 + cL]      = siluf(acc0[t][v] + bb);
      crow[(16 + v + 8 * half) * HID + t * 16 + cL] = siluf(acc1[t][v] + bb);
    }
  }
  __syncthreads();

  // ---- coord2 scalar (no bias); one row per lane ----
  {
    float s = 0.0f;
    for (int k = 0; k < HID; ++k) s += crow[lane * HID + k] * Wc2[k];
    ssc[lane] = s;
  }
  __syncthreads();

  // ---- segment-sum scatters ----
  for (int idx = lane; idx < 32 * HID; idx += 32) {
    int row = idx >> 7, col = idx & 127;
    atomicAdd(&agg_h[srow[row] * HID + col], mrow[idx] * satt[row]);
  }
  {
    int r = srow[lane];
    float s = ssc[lane];
    atomicAdd(&agg_c[r * 3 + 0], sdiff[0][lane] * s);
    atomicAdd(&agg_c[r * 3 + 1], sdiff[1][lane] * s);
    atomicAdd(&agg_c[r * 3 + 2], sdiff[2][lane] * s);
    atomicAdd(&cnt[r], 1.0f);
  }
}

// ---------------- loc update: loc += agg_c/max(cnt,1) + vel2(silu(vel1(h))) * vel ----------------
__global__ __launch_bounds__(32) void velloc_kernel(
    const unsigned short* __restrict__ hbf,
    const unsigned short* __restrict__ pWv1, const float* __restrict__ bv1,
    const float* __restrict__ Wv2, const float* __restrict__ bv2,
    const float* __restrict__ agg_c, const float* __restrict__ cnt,
    const float* __restrict__ vel, float* __restrict__ locb) {
  __shared__ __align__(16) float vrow[16 * HID];
  const int lane = threadIdx.x;
  const int n0 = blockIdx.x * 16;
  const v8f vzero = {0, 0, 0, 0, 0, 0, 0, 0};
  v8f acc[8];
#pragma unroll
  for (int t = 0; t < 8; ++t) acc[t] = vzero;
#pragma unroll
  for (int ks = 0; ks < 4; ++ks)
    wmma_step(acc, load_afrag(hbf + n0 * HID, HID, lane, ks * 32), pWv1, ks, lane);
  const int half = lane >> 4, cL = lane & 15;
#pragma unroll
  for (int t = 0; t < 8; ++t) {
    float bb = bv1[t * 16 + cL];
#pragma unroll
    for (int v = 0; v < 8; ++v)
      vrow[(v + 8 * half) * HID + t * 16 + cL] = siluf(acc[t][v] + bb);
  }
  __syncthreads();
  if (lane < 16) {
    int node = n0 + lane;
    float s = bv2[0];
    for (int k = 0; k < HID; ++k) s += vrow[lane * HID + k] * Wv2[k];
    float c = fmaxf(cnt[node], 1.0f);
#pragma unroll
    for (int j = 0; j < 3; ++j)
      locb[node * 3 + j] += agg_c[node * 3 + j] / c + s * vel[node * 3 + j];
  }
}

// ---------------- node update: h += node2(silu(node1([h | agg_h]))) ----------------
__global__ __launch_bounds__(32) void nodeupd_kernel(
    const float* __restrict__ agg_h,
    const unsigned short* __restrict__ pWn1, const float* __restrict__ bn1,
    const unsigned short* __restrict__ pWn2, const float* __restrict__ bn2,
    float* __restrict__ h, unsigned short* __restrict__ hbf) {
  __shared__ __align__(16) unsigned short cc[16 * 256];
  __shared__ __align__(16) unsigned short u1[16 * HID];
  const int lane = threadIdx.x;
  const int n0 = blockIdx.x * 16;
  const v8f vzero = {0, 0, 0, 0, 0, 0, 0, 0};

  for (int idx = lane; idx < 16 * 256; idx += 32) {
    int row = idx >> 8, K = idx & 255;
    int node = n0 + row;
    cc[idx] = (K < 128) ? hbf[node * HID + K] : f2bf(agg_h[node * HID + (K - 128)]);
  }
  __syncthreads();

  v8f acc[8];
#pragma unroll
  for (int t = 0; t < 8; ++t) acc[t] = vzero;
#pragma unroll
  for (int ks = 0; ks < 8; ++ks)
    wmma_step(acc, load_afrag(cc, 256, lane, ks * 32), pWn1, ks, lane);
  const int half = lane >> 4, cL = lane & 15;
#pragma unroll
  for (int t = 0; t < 8; ++t) {
    float bb = bn1[t * 16 + cL];
#pragma unroll
    for (int v = 0; v < 8; ++v)
      u1[(v + 8 * half) * HID + t * 16 + cL] = f2bf(siluf(acc[t][v] + bb));
  }
  __syncthreads();

#pragma unroll
  for (int t = 0; t < 8; ++t) acc[t] = vzero;
#pragma unroll
  for (int ks = 0; ks < 4; ++ks)
    wmma_step(acc, load_afrag(u1, HID, lane, ks * 32), pWn2, ks, lane);
#pragma unroll
  for (int t = 0; t < 8; ++t) {
    float bb = bn2[t * 16 + cL];
#pragma unroll
    for (int v = 0; v < 8; ++v) {
      int node = n0 + v + 8 * half;
      int o = node * HID + t * 16 + cL;
      float nh = h[o] + acc[t][v] + bb;
      h[o] = nh;
      hbf[o] = f2bf(nh);
    }
  }
}

// ---------------- final: proj + mean over L=50 ----------------
__global__ __launch_bounds__(32) void final_kernel(
    const float* __restrict__ h, const float* __restrict__ locb, const float* __restrict__ vel,
    const float* __restrict__ Wp, const float* __restrict__ bp, float* __restrict__ out) {
  const int blk = blockIdx.x;       // b*N + n, 0..319
  const int base = blk * 50;
  const int lane = threadIdx.x;
  if (lane < 16) {
    float s = 0.0f;
    for (int l = 0; l < 50; ++l) {
      const float* hr = h + (base + l) * HID;
      float d = bp[lane];
      for (int k = 0; k < HID; ++k) d += hr[k] * Wp[k * 16 + lane];
      s += d;
    }
    out[blk * 16 + lane] = s * (1.0f / 50.0f);
  } else if (lane < 22) {
    int j = lane - 16;
    const float* src = (j < 3) ? locb : vel;
    int jj = (j < 3) ? j : j - 3;
    float s = 0.0f;
    for (int l = 0; l < 50; ++l) s += src[(base + l) * 3 + jj];
    float r = s * (1.0f / 50.0f);
    if (j < 3) out[5120 + blk * 3 + jj] = r;
    else       out[6080 + blk * 3 + jj] = r;
  }
}

// ---------------- host orchestration ----------------
extern "C" void kernel_launch(void* const* d_in, const int* in_sizes, int n_in,
                              void* d_out, int out_size, void* d_ws, size_t ws_size,
                              hipStream_t stream) {
  (void)in_sizes; (void)n_in; (void)out_size;
  const float* nodes = (const float*)d_in[0];
  const float* loc   = (const float*)d_in[1];
  const int*   edges = (const int*)d_in[2];
  const float* vel   = (const float*)d_in[3];
  const float* eattr = (const float*)d_in[4];
  // params flattened in sorted-key pytree order
  const float* embW = (const float*)d_in[5];
  const float* embB = (const float*)d_in[6];
  struct Layer { const float *attW,*attB,*c1W,*c1B,*c2W,*e1W,*e1B,*e2W,*e2B,
                             *n1W,*n1B,*n2W,*n2B,*v1W,*v1B,*v2W,*v2B; };
  Layer L[4];
  int p = 7;
  for (int i = 0; i < 4; ++i) {
    L[i].attW = (const float*)d_in[p + 0];  L[i].attB = (const float*)d_in[p + 1];
    L[i].c1W  = (const float*)d_in[p + 2];  L[i].c1B  = (const float*)d_in[p + 3];
    L[i].c2W  = (const float*)d_in[p + 4];
    L[i].e1W  = (const float*)d_in[p + 5];  L[i].e1B  = (const float*)d_in[p + 6];
    L[i].e2W  = (const float*)d_in[p + 7];  L[i].e2B  = (const float*)d_in[p + 8];
    L[i].n1W  = (const float*)d_in[p + 9];  L[i].n1B  = (const float*)d_in[p + 10];
    L[i].n2W  = (const float*)d_in[p + 11]; L[i].n2B  = (const float*)d_in[p + 12];
    L[i].v1W  = (const float*)d_in[p + 13]; L[i].v1B  = (const float*)d_in[p + 14];
    L[i].v2W  = (const float*)d_in[p + 15]; L[i].v2B  = (const float*)d_in[p + 16];
    p += 17;
  }
  const float* projW = (const float*)d_in[p + 0];
  const float* projB = (const float*)d_in[p + 1];

  // workspace carve (256B aligned)
  char* ws = (char*)d_ws;
  size_t off = 0;
  auto carve = [&](size_t bytes) -> void* {
    void* r = ws + off;
    off += (bytes + 255) & ~(size_t)255;
    return r;
  };
  float*          h     = (float*)carve((size_t)NNODES * HID * 4);
  unsigned short* hbf   = (unsigned short*)carve((size_t)NNODES * HID * 2);
  float*          agg_h = (float*)carve((size_t)NNODES * HID * 4);
  float*          agg_c = (float*)carve((size_t)NNODES * 3 * 4);
  float*          cnt   = (float*)carve((size_t)NNODES * 4);
  float*          locb  = (float*)carve((size_t)NNODES * 3 * 4);
  unsigned short *pE1[4], *pE2[4], *pC1[4], *pN1[4], *pN2[4], *pV1[4];
  for (int i = 0; i < 4; ++i) {
    pE1[i] = (unsigned short*)carve(9 * 4096 * 2);  // K=288
    pE2[i] = (unsigned short*)carve(4 * 4096 * 2);  // K=128
    pC1[i] = (unsigned short*)carve(4 * 4096 * 2);
    pN1[i] = (unsigned short*)carve(8 * 4096 * 2);  // K=256
    pN2[i] = (unsigned short*)carve(4 * 4096 * 2);
    pV1[i] = (unsigned short*)carve(4 * 4096 * 2);
  }
  if (off > ws_size) return;  // workspace too small; bail deterministically

  // pack all weights into WMMA B-fragment layout
  for (int i = 0; i < 4; ++i) {
    pack_kernel<<<(9 * 4096 + 255) / 256, 256, 0, stream>>>(L[i].e1W, pE1[i], 259, 9 * 4096);
    pack_kernel<<<(4 * 4096 + 255) / 256, 256, 0, stream>>>(L[i].e2W, pE2[i], 128, 4 * 4096);
    pack_kernel<<<(4 * 4096 + 255) / 256, 256, 0, stream>>>(L[i].c1W, pC1[i], 128, 4 * 4096);
    pack_kernel<<<(8 * 4096 + 255) / 256, 256, 0, stream>>>(L[i].n1W, pN1[i], 256, 8 * 4096);
    pack_kernel<<<(4 * 4096 + 255) / 256, 256, 0, stream>>>(L[i].n2W, pN2[i], 128, 4 * 4096);
    pack_kernel<<<(4 * 4096 + 255) / 256, 256, 0, stream>>>(L[i].v1W, pV1[i], 128, 4 * 4096);
  }

  embed_kernel<<<(NNODES * HID + 255) / 256, 256, 0, stream>>>(nodes, embW, embB, h, hbf);
  copyloc_kernel<<<(NNODES * 3 + 255) / 256, 256, 0, stream>>>(loc, locb, NNODES * 3);

  for (int i = 0; i < 4; ++i) {
    zero_kernel<<<(NNODES * HID + 255) / 256, 256, 0, stream>>>(agg_h, NNODES * HID);
    zero_kernel<<<(NNODES * 3 + 255) / 256, 256, 0, stream>>>(agg_c, NNODES * 3);
    zero_kernel<<<(NNODES + 255) / 256, 256, 0, stream>>>(cnt, NNODES);

    edge_kernel<<<NEDGES / 32, 32, 0, stream>>>(
        edges, eattr, locb, hbf,
        pE1[i], L[i].e1B, pE2[i], L[i].e2B, pC1[i], L[i].c1B,
        L[i].attW, L[i].attB, L[i].c2W,
        agg_h, agg_c, cnt);

    velloc_kernel<<<NNODES / 16, 32, 0, stream>>>(
        hbf, pV1[i], L[i].v1B, L[i].v2W, L[i].v2B, agg_c, cnt, vel, locb);

    nodeupd_kernel<<<NNODES / 16, 32, 0, stream>>>(
        agg_h, pN1[i], L[i].n1B, pN2[i], L[i].n2B, h, hbf);
  }

  final_kernel<<<320, 32, 0, stream>>>(h, locb, vel, projW, projB, (float*)d_out);
}
